// TimeAwareFeatViT_7035156431003
// MI455X (gfx1250) — compile-verified, matrix operand order checked by reading, not run on hardware
//
#include <hip/hip_runtime.h>
#include <math.h>

// ---------------- types / constants ----------------
typedef __attribute__((ext_vector_type(16))) __bf16 v16bf;
typedef __attribute__((ext_vector_type(8)))  float  v8f;
typedef unsigned short u16;
typedef unsigned int   u32;
typedef long long      i64;

#define BATCH  16
#define TOKENS 512
#define FDIM   1024
#define DIM    768
#define DEPTH  12
#define HEADS  12
#define DHEAD  64
#define MLPD   3072
#define NCLS   2
#define MTOK   (BATCH * TOKENS)   // 8192 rows in the residual stream

// ---- CDNA5 async global->LDS path (probe-safe via __has_builtin) ----
#if defined(__AMDGCN__) && __has_builtin(__builtin_amdgcn_global_load_async_to_lds_b128)
#define HAS_ASYNC 1
typedef int v4i_ __attribute__((ext_vector_type(4)));
typedef __attribute__((address_space(1))) v4i_ as1_v4i;   // global (prints as __device__)
typedef __attribute__((address_space(3))) v4i_ as3_v4i;   // LDS    (prints as __shared__)
#define ASYNC_CP16(gp, lp) \
  __builtin_amdgcn_global_load_async_to_lds_b128((as1_v4i*)(gp), (as3_v4i*)(lp), 0, 0)
#if __has_builtin(__builtin_amdgcn_s_wait_asynccnt)
#define WAIT_ASYNC(n) __builtin_amdgcn_s_wait_asynccnt(n)
#else
#define WAIT_ASYNC(n) asm volatile("s_wait_asynccnt %0" :: "n"(n) : "memory")
#endif
#else
#define HAS_ASYNC 0
#endif

// ---------------- bf16 helpers ----------------
__device__ __forceinline__ u16 f2bf(float f) {
  u32 u = __float_as_uint(f);
  u32 r = u + 0x7FFFu + ((u >> 16) & 1u);   // round-to-nearest-even
  return (u16)(r >> 16);
}
__device__ __forceinline__ float bf2f(u16 h) {
  return __uint_as_float(((u32)h) << 16);
}
__device__ __forceinline__ float gelu_f(float x) {
  const float c0 = 0.7978845608028654f;     // sqrt(2/pi)
  return 0.5f * x * (1.0f + tanhf(c0 * (x + 0.044715f * x * x * x)));
}

// ---------------- block reductions (blockDim == 256) ----------------
__device__ __forceinline__ float blk_sum(float v, float* red) {
  int t = threadIdx.x;
  red[t] = v; __syncthreads();
  #pragma unroll
  for (int s = 128; s > 0; s >>= 1) { if (t < s) red[t] += red[t + s]; __syncthreads(); }
  float r = red[0]; __syncthreads();
  return r;
}
__device__ __forceinline__ float blk_max(float v, float* red) {
  int t = threadIdx.x;
  red[t] = v; __syncthreads();
  #pragma unroll
  for (int s = 128; s > 0; s >>= 1) { if (t < s) red[t] = fmaxf(red[t], red[t + s]); __syncthreads(); }
  float r = red[0]; __syncthreads();
  return r;
}

// ---------------- WMMA GEMM ----------------
// D[M,N] = epilogue( alpha * A[M,K] @ B^T ), A row-major [M,K], B row-major [N,K]
// (every B operand is pre-transposed so both tiles are contiguous 64B strips).
// bf16 inputs, f32 WMMA accumulate. Double-buffered LDS, async copies when available.
// Batched via blockIdx.z with (outer,inner) = (z/zdiv, z%zdiv) element offsets.
#define BM   128
#define BN   128
#define BKK  32
#define SBK  40   // LDS row stride in u16 (32 + 8 pad; 80B keeps 16B alignment)
#define SBKU 20   // same, in u32
#define BUFU (BM * SBK / 2)  // u32 per LDS buffer

__device__ __forceinline__ void wmma_tile(const u32* AsU, const u32* BsU,
                                          int waveM, int waveN, int hh, int ll,
                                          v8f acc[2][4]) {
  union Frag { v16bf v; u32 u[8]; };
  Frag fa[2], fb[4];
  #pragma unroll
  for (int tm = 0; tm < 2; ++tm) {
    const u32* p = AsU + (waveM + tm * 16 + ll) * SBKU;
    #pragma unroll
    for (int v = 0; v < 8; ++v) fa[tm].u[v] = p[(v & 3) + hh * 4 + ((v & 4) ? 8 : 0)];
  }
  #pragma unroll
  for (int tn = 0; tn < 4; ++tn) {
    const u32* p = BsU + (waveN + tn * 16 + ll) * SBKU;
    #pragma unroll
    for (int v = 0; v < 8; ++v) fb[tn].u[v] = p[(v & 3) + hh * 4 + ((v & 4) ? 8 : 0)];
  }
  #pragma unroll
  for (int tm = 0; tm < 2; ++tm)
    #pragma unroll
    for (int tn = 0; tn < 4; ++tn)
      acc[tm][tn] = __builtin_amdgcn_wmma_f32_16x16x32_bf16(
          false, fa[tm].v, false, fb[tn].v, (short)0, acc[tm][tn], false, false);
}

template<bool NGUARD, bool BIAS, bool RESID, bool DOGELU, bool OUTBF16, bool ROWB>
__global__ __launch_bounds__(256)
void gemm_wmma(const u16* __restrict__ A, const u16* __restrict__ B, void* Dp,
               const float* __restrict__ bias, const float* resid,
               const float* __restrict__ rowBias,
               int M, int N, int K, int lda, int ldb, int ldd,
               int zdiv, i64 offAo, i64 offAi, i64 offBo, i64 offBi, i64 offDo, i64 offDi,
               int rbDiv, float alpha)
{
  __shared__ __align__(16) u16 As[2 * BM * SBK];
  __shared__ __align__(16) u16 Bs[2 * BN * SBK];

  const int tid  = threadIdx.x;
  const int lane = tid & 31;
  const int wid  = tid >> 5;
  const int z    = blockIdx.z;
  const int zo   = z / zdiv;
  const int zi   = z - zo * zdiv;

  const u16* Ag = A + (i64)zo * offAo + (i64)zi * offAi;
  const u16* Bg = B + (i64)zo * offBo + (i64)zi * offBi;
  const i64 dOff = (i64)zo * offDo + (i64)zi * offDi;

  const int blockM = blockIdx.y * BM;
  const int blockN = blockIdx.x * BN;
  const int waveM  = (wid >> 1) * 32;
  const int waveN  = (wid & 1) * 64;
  const int hh     = lane >> 4;
  const int ll     = lane & 15;

  // staging geometry: thread handles rows rr and rr+64 of both tiles, 16B each
  const int rr = tid >> 2;          // 0..63
  const int ss = (tid & 3) * 8;     // u16 offset 0,8,16,24
  const u16* gA0 = Ag + (i64)(blockM + rr) * lda + ss;
  const u16* gA1 = gA0 + (i64)64 * lda;
  int nb0 = blockN + rr, nb1 = nb0 + 64;
  if (NGUARD) {                      // clamp (dup rows); garbage cols never stored
    nb0 = (nb0 < N) ? nb0 : (N - 1);
    nb1 = (nb1 < N) ? nb1 : (N - 1);
  }
  const u16* gB0 = Bg + (i64)nb0 * ldb + ss;
  const u16* gB1 = Bg + (i64)nb1 * ldb + ss;
  u16* lA0 = As + rr * SBK + ss;  u16* lA1 = lA0 + 64 * SBK;
  u16* lB0 = Bs + rr * SBK + ss;  u16* lB1 = lB0 + 64 * SBK;
  const int bufU16 = BM * SBK;     // u16 per double-buffer slot

  v8f acc[2][4];
  #pragma unroll
  for (int a_ = 0; a_ < 2; ++a_)
    #pragma unroll
    for (int b_ = 0; b_ < 4; ++b_)
      #pragma unroll
      for (int e = 0; e < 8; ++e) acc[a_][b_][e] = 0.0f;

  const int NT = K / BKK;
#if HAS_ASYNC
  ASYNC_CP16(gA0, lA0); ASYNC_CP16(gA1, lA1);
  ASYNC_CP16(gB0, lB0); ASYNC_CP16(gB1, lB1);
  for (int kt = 0; kt < NT; ++kt) {
    const int buf = kt & 1;
    if (kt + 1 < NT) {
      const i64 ko = (i64)(kt + 1) * BKK;
      const int bo = (buf ^ 1) * bufU16;
      ASYNC_CP16(gA0 + ko, lA0 + bo); ASYNC_CP16(gA1 + ko, lA1 + bo);
      ASYNC_CP16(gB0 + ko, lB0 + bo); ASYNC_CP16(gB1 + ko, lB1 + bo);
      WAIT_ASYNC(4);                 // tile kt done (in-order), kt+1 in flight
    } else {
      WAIT_ASYNC(0);
    }
    __syncthreads();
    wmma_tile((const u32*)As + buf * BUFU, (const u32*)Bs + buf * BUFU,
              waveM, waveN, hh, ll, acc);
    __syncthreads();
  }
#else
  uint4 ra0 = *(const uint4*)gA0, ra1 = *(const uint4*)gA1;
  uint4 rb0 = *(const uint4*)gB0, rb1 = *(const uint4*)gB1;
  for (int kt = 0; kt < NT; ++kt) {
    const int buf = kt & 1;
    const int bo = buf * bufU16;
    *(uint4*)(lA0 + bo) = ra0; *(uint4*)(lA1 + bo) = ra1;
    *(uint4*)(lB0 + bo) = rb0; *(uint4*)(lB1 + bo) = rb1;
    if (kt + 1 < NT) {               // prefetch next tile into regs (overlaps WMMA)
      const i64 ko = (i64)(kt + 1) * BKK;
      ra0 = *(const uint4*)(gA0 + ko); ra1 = *(const uint4*)(gA1 + ko);
      rb0 = *(const uint4*)(gB0 + ko); rb1 = *(const uint4*)(gB1 + ko);
    }
    __syncthreads();
    wmma_tile((const u32*)As + buf * BUFU, (const u32*)Bs + buf * BUFU,
              waveM, waveN, hh, ll, acc);
    __syncthreads();
  }
#endif

  // ---- epilogue ----
  const float* rbp = ROWB ? (rowBias + (i64)(z / rbDiv) * M) : nullptr;
  #pragma unroll
  for (int tm = 0; tm < 2; ++tm) {
    #pragma unroll
    for (int tn = 0; tn < 4; ++tn) {
      int nCol = blockN + waveN + tn * 16 + ll;
      if (NGUARD && nCol >= N) continue;
      #pragma unroll
      for (int v = 0; v < 8; ++v) {
        int m = blockM + waveM + tm * 16 + v + hh * 8;  // C/D: M = vgpr + half*8
        float val = acc[tm][tn][v] * alpha;
        if (ROWB)   val += rbp[m];
        if (BIAS)   val += bias[nCol];
        if (DOGELU) val = gelu_f(val);
        i64 idx = dOff + (i64)m * ldd + nCol;
        if (RESID)  val += resid[idx];
        if (OUTBF16) ((u16*)Dp)[idx] = f2bf(val);
        else         ((float*)Dp)[idx] = val;
      }
    }
  }
}

// ---------------- elementwise / reduction kernels ----------------
__global__ void cvt_bf16_k(const float* __restrict__ in, u16* __restrict__ out, i64 n) {
  i64 i = (i64)blockIdx.x * blockDim.x + threadIdx.x;
  i64 stride = (i64)gridDim.x * blockDim.x;
  for (; i < n; i += stride) out[i] = f2bf(in[i]);
}

// per-layer transpose + convert: in (depth-batched) [Kd x Nd] f32 -> out [Nd x Kd] bf16
__global__ void cvtT_k(const float* __restrict__ in, u16* __restrict__ out,
                       int Kd, int Nd, i64 total) {
  const i64 KN = (i64)Kd * Nd;
  i64 i = (i64)blockIdx.x * blockDim.x + threadIdx.x;
  i64 stride = (i64)gridDim.x * blockDim.x;
  for (; i < total; i += stride) {
    i64 l = i / KN, o = i - l * KN;
    int n = (int)(o / Kd), k = (int)(o - (i64)n * Kd);
    out[i] = f2bf(in[l * KN + (i64)k * Nd + n]);
  }
}

// vt[(z*DHEAD + d)*TOKENS + j] = V[b, j, h, d]  (V inside packed qkv)
__global__ void vtrans_k(const u16* __restrict__ qkv, u16* __restrict__ vt) {
  i64 i = (i64)blockIdx.x * blockDim.x + threadIdx.x;
  if (i >= (i64)BATCH * HEADS * DHEAD * TOKENS) return;
  int j = (int)(i & (TOKENS - 1));
  i64 t1 = i >> 9;                 // z*DHEAD + d
  int d = (int)(t1 & (DHEAD - 1));
  int zz = (int)(t1 >> 6);
  int b = zz / HEADS, h = zz - b * HEADS;
  vt[i] = qkv[((i64)(b * TOKENS + j)) * (3 * DIM) + 2 * DIM + h * DHEAD + d];
}

__global__ void rowbias_k(const float* __restrict__ times, float* __restrict__ rb) {
  int i = blockIdx.x * blockDim.x + threadIdx.x;
  if (i >= BATCH * TOKENS) return;
  int b = i / TOKENS;
  rb[i] = fabsf(times[b * TOKENS] - times[i]);
}

__global__ __launch_bounds__(256)
void ln_to_bf16(const float* __restrict__ x, const float* __restrict__ w,
                const float* __restrict__ b, u16* __restrict__ out) {
  __shared__ float red[256];
  const i64 row = blockIdx.x;
  const float* xr = x + row * DIM;
  const int t = threadIdx.x;
  float v[3]; float s = 0.f;
  #pragma unroll
  for (int i = 0; i < 3; ++i) { v[i] = xr[t + i * 256]; s += v[i]; }
  float mean = blk_sum(s, red) * (1.0f / DIM);
  float q = 0.f;
  #pragma unroll
  for (int i = 0; i < 3; ++i) { float d = v[i] - mean; q += d * d; }
  float var = blk_sum(q, red) * (1.0f / DIM);
  float inv = rsqrtf(var + 1e-5f);
  u16* orow = out + row * DIM;
  #pragma unroll
  for (int i = 0; i < 3; ++i) {
    int c = t + i * 256;
    orow[c] = f2bf((v[i] - mean) * inv * w[c] + b[c]);
  }
}

__global__ __launch_bounds__(256)
void softmax_bf16(u16* __restrict__ sc) {
  __shared__ float red[256];
  const i64 row = blockIdx.x;
  u16* sr = sc + row * TOKENS;
  const int t = threadIdx.x;
  float v0 = bf2f(sr[t]), v1 = bf2f(sr[t + 256]);
  float mx = blk_max(fmaxf(v0, v1), red);
  float e0 = __expf(v0 - mx), e1 = __expf(v1 - mx);
  float inv = 1.0f / blk_sum(e0 + e1, red);
  sr[t]       = f2bf(e0 * inv);
  sr[t + 256] = f2bf(e1 * inv);
}

__global__ void meanpool_k(const float* __restrict__ x, float* __restrict__ out) {
  int i = blockIdx.x * blockDim.x + threadIdx.x;
  if (i >= BATCH * DIM) return;
  int b = i / DIM, d = i - b * DIM;
  float s = 0.f;
  for (int tk = 0; tk < TOKENS; ++tk) s += x[((i64)b * TOKENS + tk) * DIM + d];
  out[i] = s * (1.0f / TOKENS);
}

__global__ __launch_bounds__(256)
void head_k(const float* __restrict__ pooled,
            const float* __restrict__ hw, const float* __restrict__ hb,
            const float* __restrict__ w1, const float* __restrict__ b1,
            const float* __restrict__ w2, const float* __restrict__ b2,
            float* __restrict__ out) {
  __shared__ float red[256];
  __shared__ float xn[DIM];
  __shared__ float hid[DIM];
  const int b = blockIdx.x, t = threadIdx.x;
  const float* xr = pooled + (i64)b * DIM;
  float v[3]; float s = 0.f;
  #pragma unroll
  for (int i = 0; i < 3; ++i) { v[i] = xr[t + i * 256]; s += v[i]; }
  float mean = blk_sum(s, red) * (1.0f / DIM);
  float q = 0.f;
  #pragma unroll
  for (int i = 0; i < 3; ++i) { float d = v[i] - mean; q += d * d; }
  float var = blk_sum(q, red) * (1.0f / DIM);
  float inv = rsqrtf(var + 1e-5f);
  #pragma unroll
  for (int i = 0; i < 3; ++i) {
    int c = t + i * 256;
    xn[c] = (v[i] - mean) * inv * hw[c] + hb[c];
  }
  __syncthreads();
  for (int j = t; j < DIM; j += 256) {
    float a = 0.f;
    for (int d = 0; d < DIM; ++d) a += xn[d] * w1[(i64)d * DIM + j];
    hid[j] = a + b1[j];
  }
  __syncthreads();
  for (int c = 0; c < NCLS; ++c) {
    float a = 0.f;
    for (int j = t; j < DIM; j += 256) a += hid[j] * w2[(i64)j * NCLS + c];
    float tot = blk_sum(a, red);
    if (t == 0) out[b * NCLS + c] = tot + b2[c];
  }
}

// ---------------- driver ----------------
extern "C" void kernel_launch(void* const* d_in, const int* in_sizes, int n_in,
                              void* d_out, int out_size, void* d_ws, size_t ws_size,
                              hipStream_t stream) {
  (void)in_sizes; (void)n_in; (void)out_size; (void)ws_size;
  const float* feat  = (const float*)d_in[0];
  const float* times = (const float*)d_in[1];
  const float* emb_w = (const float*)d_in[2];
  const float* emb_b = (const float*)d_in[3];
  const float* ln1_w = (const float*)d_in[4];
  const float* ln1_b = (const float*)d_in[5];
  const float* w_qkv = (const float*)d_in[6];
  const float* w_o   = (const float*)d_in[7];
  const float* b_o   = (const float*)d_in[8];
  const float* ln2_w = (const float*)d_in[9];
  const float* ln2_b = (const float*)d_in[10];
  const float* w1    = (const float*)d_in[11];
  const float* b1    = (const float*)d_in[12];
  const float* w2    = (const float*)d_in[13];
  const float* b2    = (const float*)d_in[14];
  const float* hln_w = (const float*)d_in[15];
  const float* hln_b = (const float*)d_in[16];
  const float* hw1   = (const float*)d_in[17];
  const float* hb1   = (const float*)d_in[18];
  const float* hw2   = (const float*)d_in[19];
  const float* hb2   = (const float*)d_in[20];

  char* wsp = (char*)d_ws; size_t off = 0;
  auto alloc = [&](size_t bytes) -> void* {
    void* p = wsp + off;
    off += (bytes + 255) & ~(size_t)255;
    return p;
  };
  u16*   featB  = (u16*)alloc((size_t)BATCH * TOKENS * FDIM * 2);
  u16*   embwT  = (u16*)alloc((size_t)FDIM * DIM * 2);                 // [DIM, FDIM]
  u16*   wqkvT  = (u16*)alloc((size_t)DEPTH * DIM * 3 * DIM * 2);      // [3*DIM, DIM] per layer
  u16*   woT    = (u16*)alloc((size_t)DEPTH * DIM * DIM * 2);          // [DIM, DIM]
  u16*   w1T    = (u16*)alloc((size_t)DEPTH * DIM * MLPD * 2);         // [MLPD, DIM]
  u16*   w2T    = (u16*)alloc((size_t)DEPTH * MLPD * DIM * 2);         // [DIM, MLPD]
  float* x      = (float*)alloc((size_t)MTOK * DIM * 4);
  u16*   xn     = (u16*)alloc((size_t)MTOK * DIM * 2);
  u16*   qkv    = (u16*)alloc((size_t)MTOK * 3 * DIM * 2);
  u16*   vt     = (u16*)alloc((size_t)BATCH * HEADS * DHEAD * TOKENS * 2);
  u16*   sc     = (u16*)alloc((size_t)BATCH * HEADS * TOKENS * TOKENS * 2);
  u16*   attno  = (u16*)alloc((size_t)MTOK * DIM * 2);
  u16*   hmlp   = (u16*)alloc((size_t)MTOK * MLPD * 2);
  float* rb     = (float*)alloc((size_t)BATCH * TOKENS * 4);
  float* pooled = (float*)alloc((size_t)BATCH * DIM * 4);

  auto grd = [](i64 n) { i64 b = (n + 255) / 256; return (int)(b > 262144 ? 262144 : b); };

  cvt_bf16_k<<<grd((i64)BATCH * TOKENS * FDIM), 256, 0, stream>>>(
      feat, featB, (i64)BATCH * TOKENS * FDIM);
  cvtT_k<<<grd((i64)FDIM * DIM), 256, 0, stream>>>(emb_w, embwT, FDIM, DIM, (i64)FDIM * DIM);
  cvtT_k<<<grd((i64)DEPTH * DIM * 3 * DIM), 256, 0, stream>>>(
      w_qkv, wqkvT, DIM, 3 * DIM, (i64)DEPTH * DIM * 3 * DIM);
  cvtT_k<<<grd((i64)DEPTH * DIM * DIM), 256, 0, stream>>>(
      w_o, woT, DIM, DIM, (i64)DEPTH * DIM * DIM);
  cvtT_k<<<grd((i64)DEPTH * DIM * MLPD), 256, 0, stream>>>(
      w1, w1T, DIM, MLPD, (i64)DEPTH * DIM * MLPD);
  cvtT_k<<<grd((i64)DEPTH * MLPD * DIM), 256, 0, stream>>>(
      w2, w2T, MLPD, DIM, (i64)DEPTH * MLPD * DIM);

  rowbias_k<<<(BATCH * TOKENS + 255) / 256, 256, 0, stream>>>(times, rb);

  // x = feat @ emb_w + emb_b
  gemm_wmma<false, true, false, false, false, false>
      <<<dim3(DIM / 128, MTOK / 128, 1), 256, 0, stream>>>(
      featB, embwT, x, emb_b, nullptr, nullptr,
      MTOK, DIM, FDIM, FDIM, FDIM, DIM,
      1, 0, 0, 0, 0, 0, 0, 1, 1.0f);

  const i64 TT = (i64)TOKENS * TOKENS;
  for (int l = 0; l < DEPTH; ++l) {
    ln_to_bf16<<<MTOK, 256, 0, stream>>>(x, ln1_w + l * DIM, ln1_b + l * DIM, xn);

    // qkv = xn @ w_qkv[l]
    gemm_wmma<false, false, false, false, true, false>
        <<<dim3(3 * DIM / 128, MTOK / 128, 1), 256, 0, stream>>>(
        xn, wqkvT + (i64)l * DIM * 3 * DIM, qkv, nullptr, nullptr, nullptr,
        MTOK, 3 * DIM, DIM, DIM, DIM, 3 * DIM,
        1, 0, 0, 0, 0, 0, 0, 1, 1.0f);

    vtrans_k<<<(int)(((i64)BATCH * HEADS * DHEAD * TOKENS + 255) / 256), 256, 0, stream>>>(qkv, vt);

    // scores = Q @ K^T * scale + R[b,i]  (K rows are already [N,K] layout)
    gemm_wmma<false, false, false, false, true, true>
        <<<dim3(TOKENS / 128, TOKENS / 128, BATCH * HEADS), 256, 0, stream>>>(
        qkv, qkv + DIM, sc, nullptr, nullptr, rb,
        TOKENS, TOKENS, DHEAD, 3 * DIM, 3 * DIM, TOKENS,
        HEADS,
        (i64)TOKENS * 3 * DIM, DHEAD,
        (i64)TOKENS * 3 * DIM, DHEAD,
        (i64)HEADS * TT, TT,
        HEADS, 0.125f);

    softmax_bf16<<<BATCH * HEADS * TOKENS, 256, 0, stream>>>(sc);

    // o[b,i,h,:] = P @ V   (B = V^T per (b,h); N=64 -> clamped rows + store guard)
    gemm_wmma<true, false, false, false, true, false>
        <<<dim3(1, TOKENS / 128, BATCH * HEADS), 256, 0, stream>>>(
        sc, vt, attno, nullptr, nullptr, nullptr,
        TOKENS, DHEAD, TOKENS, TOKENS, TOKENS, DIM,
        HEADS,
        (i64)HEADS * TT, TT,
        (i64)HEADS * DHEAD * TOKENS, (i64)DHEAD * TOKENS,
        (i64)TOKENS * DIM, DHEAD,
        1, 1.0f);

    // x += o @ w_o[l] + b_o[l]
    gemm_wmma<false, true, true, false, false, false>
        <<<dim3(DIM / 128, MTOK / 128, 1), 256, 0, stream>>>(
        attno, woT + (i64)l * DIM * DIM, x, b_o + l * DIM, x, nullptr,
        MTOK, DIM, DIM, DIM, DIM, DIM,
        1, 0, 0, 0, 0, 0, 0, 1, 1.0f);

    ln_to_bf16<<<MTOK, 256, 0, stream>>>(x, ln2_w + l * DIM, ln2_b + l * DIM, xn);

    // h = gelu(xn @ w1[l] + b1[l])
    gemm_wmma<false, true, false, true, true, false>
        <<<dim3(MLPD / 128, MTOK / 128, 1), 256, 0, stream>>>(
        xn, w1T + (i64)l * DIM * MLPD, hmlp, b1 + l * MLPD, nullptr, nullptr,
        MTOK, MLPD, DIM, DIM, DIM, MLPD,
        1, 0, 0, 0, 0, 0, 0, 1, 1.0f);

    // x += h @ w2[l] + b2[l]
    gemm_wmma<false, true, true, false, false, false>
        <<<dim3(DIM / 128, MTOK / 128, 1), 256, 0, stream>>>(
        hmlp, w2T + (i64)l * MLPD * DIM, x, b2 + l * DIM, x, nullptr,
        MTOK, DIM, MLPD, MLPD, MLPD, DIM,
        1, 0, 0, 0, 0, 0, 0, 1, 1.0f);
  }

  meanpool_k<<<(BATCH * DIM + 255) / 256, 256, 0, stream>>>(x, pooled);
  head_k<<<BATCH, 256, 0, stream>>>(pooled, hln_w, hln_b, hw1, hb1, hw2, hb2,
                                    (float*)d_out);
}
